// lora_quantizer_module_1408749273623
// MI455X (gfx1250) — compile-verified, matrix operand order checked by reading, not run on hardware
//
#include <hip/hip_runtime.h>
#include <stdint.h>

typedef __attribute__((ext_vector_type(16))) _Float16     v16h;
typedef __attribute__((ext_vector_type(8)))  float        v8f;
typedef __attribute__((ext_vector_type(4)))  unsigned int v4u;

#define OUT_DIM 8192
#define R_DIM   64
#define IN_DIM  8192

// -----------------------------------------------------------------------------
// Kernel 1: dequantize int64 codebook assignments -> f16 operands in d_ws.
//   Ah : [OUT_DIM][R_DIM] row-major f16   (A[m][k])
//   BTh: [IN_DIM][R_DIM]  row-major f16   (BT[n][k] = B[k][n]) so both WMMA
//        fragments load with the same per-lane K-contiguous 16B pattern.
// Total 2 MB -> stays resident in L2 for the whole GEMM.
// -----------------------------------------------------------------------------
__global__ __launch_bounds__(256)
void deq_kernel(const long long* __restrict__ Aasg,
                const long long* __restrict__ Basg,
                const float* __restrict__ Acb,
                const float* __restrict__ Bcb,
                _Float16* __restrict__ Ah,
                _Float16* __restrict__ BTh) {
  const int i = blockIdx.x * 256 + threadIdx.x;   // 0 .. 524287 (exact grid)

  // A: keep row-major (coalesced load + coalesced store)
  {
    const int idx = ((int)Aasg[i]) & 15;
    Ah[i] = (_Float16)Acb[idx];
  }
  // B: [R,IN] row-major -> transpose into BT[n*R + k]
  {
    const int k   = i >> 13;              // i / IN_DIM
    const int n   = i & (IN_DIM - 1);     // i % IN_DIM
    const int idx = ((int)Basg[i]) & 15;
    BTh[n * R_DIM + k] = (_Float16)Bcb[idx];
  }
}

// -----------------------------------------------------------------------------
// Kernel 2: low-rank WMMA GEMM. Wave32; each wave computes a 16(M) x 64(N)
// output tile with 8 x v_wmma_f32_16x16x32_f16 (K = 64 = 2 k-steps x 4 tiles).
// Block = 8 waves -> 16 x 512 outputs per block.
//
// 16-bit A-fragment layout (ISA 7.12.2), for k-step base ks:
//   lane<16 (row M=lane)    : halves 0-7 = K ks..ks+7,  halves 8-15 = K ks+16..ks+23
//   lane>=16 (row M=lane-16): halves 0-7 = K ks+8..+15, halves 8-15 = K ks+24..+31
// B-fragment uses the same pattern with N=lane&15 on BT (K-contiguous rows).
// C/D layout: VGPR i -> M = i + 8*(lane>=16), N = lane&15.
// -----------------------------------------------------------------------------
union FragU { v4u u2[2]; v16h h; };

__global__ __launch_bounds__(256)
void lowrank_wmma_kernel(const _Float16* __restrict__ Ah,
                         const _Float16* __restrict__ BTh,
                         float* __restrict__ C) {
  const int lane     = threadIdx.x & 31;
  const int wave     = threadIdx.x >> 5;
  const int half_sel = lane >> 4;              // 0: lanes 0-15, 1: lanes 16-31
  const int row      = lane & 15;

  const int m_base = blockIdx.y << 4;                       // 16 rows per block
  const int n_base = (blockIdx.x << 9) + (wave << 6);       // 64 cols per wave

  // per-lane base pointers (units: halves; all 16-byte aligned)
  const _Float16* arow = Ah  + (size_t)(m_base + row) * R_DIM + half_sel * 8;

  v8f acc[4] = {};

#pragma unroll
  for (int ks = 0; ks < R_DIM; ks += 32) {
    FragU fa;
    fa.u2[0] = *(const v4u*)(arow + ks);        // K ks..ks+7   (or +8..+15)
    fa.u2[1] = *(const v4u*)(arow + ks + 16);   // K ks+16..+23 (or +24..+31)

#pragma unroll
    for (int t = 0; t < 4; ++t) {
      const _Float16* bcol =
          BTh + (size_t)(n_base + t * 16 + row) * R_DIM + half_sel * 8 + ks;
      FragU fb;
      fb.u2[0] = *(const v4u*)(bcol);
      fb.u2[1] = *(const v4u*)(bcol + 16);

      // D = A x B + C, f32 accumulate
      acc[t] = __builtin_amdgcn_wmma_f32_16x16x32_f16(
          /*neg_a=*/false, fa.h, /*neg_b=*/false, fb.h,
          /*c_mod=*/(short)0, acc[t], /*reuse_a=*/false, /*reuse_b=*/false);
    }
  }

  // Store 16x64 f32 tile. Non-temporal: 256 MB streamed once; keep it out of
  // L2 so the 2 MB of f16 operands stay resident across their 512x reuse.
#pragma unroll
  for (int t = 0; t < 4; ++t) {
    const int col = n_base + t * 16 + row;
#pragma unroll
    for (int i = 0; i < 8; ++i) {
      const int m = m_base + i + half_sel * 8;
      __builtin_nontemporal_store(acc[t][i], &C[(size_t)m * IN_DIM + col]);
    }
  }
}

// -----------------------------------------------------------------------------
// Harness entry point.
//   d_in[0]: A_assignments int64 [8192*64]
//   d_in[1]: B_assignments int64 [64*8192]
//   d_in[2]: A_codebook f32 [16]
//   d_in[3]: B_codebook f32 [16]
//   d_out  : f32 [8192*8192]
//   d_ws   : >= 2 MB scratch for f16 A / B^T
// -----------------------------------------------------------------------------
extern "C" void kernel_launch(void* const* d_in, const int* in_sizes, int n_in,
                              void* d_out, int out_size, void* d_ws, size_t ws_size,
                              hipStream_t stream) {
  (void)in_sizes; (void)n_in; (void)out_size; (void)ws_size;

  const long long* Aasg = (const long long*)d_in[0];
  const long long* Basg = (const long long*)d_in[1];
  const float*     Acb  = (const float*)d_in[2];
  const float*     Bcb  = (const float*)d_in[3];
  float*           Cout = (float*)d_out;

  _Float16* Ah  = (_Float16*)d_ws;                 // 8192*64  halves = 1 MB
  _Float16* BTh = Ah + (size_t)OUT_DIM * R_DIM;    // 64*8192  halves = 1 MB

  // Stage 1: dequantize (exactly 524288 elements for both A and B)
  deq_kernel<<<dim3((OUT_DIM * R_DIM) / 256), 256, 0, stream>>>(
      Aasg, Basg, Acb, Bcb, Ah, BTh);

  // Stage 2: WMMA GEMM. Grid: x = 8192/512 N-chunks, y = 8192/16 M-chunks.
  dim3 grid(IN_DIM / 512, OUT_DIM / 16);
  lowrank_wmma_kernel<<<grid, 256, 0, stream>>>(Ah, BTh, Cout);
}
